// MambaSSMBlock_46033459479199
// MI455X (gfx1250) — matrix operand order repaired and checked by reading, not compiled, not run on hardware
//
#include <hip/hip_runtime.h>
#include <hip/hip_bf16.h>

// ---------------- types ----------------
typedef __bf16 bf16_t;
typedef __attribute__((ext_vector_type(16))) __bf16 v16bf;
typedef __attribute__((ext_vector_type(8)))  __bf16 v8bf;
typedef __attribute__((ext_vector_type(8)))  float  v8f;

// ---------------- problem constants ----------------
#define D_MODEL 1024
#define D_STATE 16
#define D_CONV  4
#define D_INNER 2048
#define DT_RANK 64
#define BATCH   2
#define SEQL    1024
#define NROW    (BATCH * SEQL)   // 2048 token rows

__device__ __forceinline__ float sigmoidf_(float x) { return 1.0f / (1.0f + __expf(-x)); }

// =====================================================================
// Generic tiled GEMM: C[M,N] = A[M,K] * W[N,K]^T   (row-major, f32 in/out)
// bf16 WMMA (v_wmma_f32_16x16x32_bf16), f32 accumulate.
// Block = 128 threads (4 waves, wave32). Tile: BM=64, BN=128, BK=32.
// Wave w owns M-subtile w (16 rows) x all 8 N-subtiles -> 8 WMMA / K-step.
// LDS tiles are stored in WMMA *fragment order* so each lane's fragment
// load is a single contiguous 32-byte LDS read.
// Requires: M % 64 == 0, K % 32 == 0, N % 16 == 0 (true for all launches).
// =====================================================================
__launch_bounds__(128)
__global__ void gemm_wmma_bf16(const float* __restrict__ A,
                               const float* __restrict__ W,
                               float* __restrict__ C,
                               int M, int N, int K,
                               int lda, int ldw, int ldc)
{
    __shared__ __align__(32) bf16_t Abuf[2048];  // 4 msub * 32 lanes * 16 bf16 (4 KB)
    __shared__ __align__(32) bf16_t Bbuf[4096];  // 8 nsub * 32 lanes * 16 bf16 (8 KB)

    const int t    = threadIdx.x;
    const int lane = t & 31;
    const int w    = t >> 5;
    const int m0   = blockIdx.y * 64;
    const int n0   = blockIdx.x * 128;

    v8f acc[8] = {};

    // ---- A-fill mapping: thread -> (row within tile, 16-wide k group) ----
    // 16-bit A-matrix 16x32 fragment layout (ISA 7.12.2):
    //   lanes 0-15 : j 0..7 -> K 0..7,  j 8..15 -> K 16..23
    //   lanes 16-31: j 0..7 -> K 8..15, j 8..15 -> K 24..31
    // => contiguous k-run [kg, kg+15] splits into two 8-elem contiguous
    //    j-runs at lanes (mrow) and (mrow+16), jbase = kg/2.
    const int am     = t >> 1;        // 0..63 tile row
    const int akg    = (t & 1) * 16;  // k group 0 or 16
    const int amsub  = am >> 4;
    const int amrow  = am & 15;
    const int ajbase = akg >> 1;      // 0 or 8

    // ---- B-fill mapping: B fragment lane = N (lane%16), element j = K (table order) ----
    // Each thread fills fragment lane (t&31) of two n-subtile groups: (t>>5), (t>>5)+4.
    const int bk = t & 31;
    const int bg = t >> 5;            // wave-uniform -> group guards are s_cbranch

    for (int k0 = 0; k0 < K; k0 += 32) {
        // ---- stage A tile (64 x 32), f32 -> bf16, fragment-ordered ----
        {
            const float* ap = A + (size_t)(m0 + am) * lda + (k0 + akg);
            v8bf lo, hi;
#pragma unroll
            for (int i = 0; i < 8; ++i) lo[i] = (bf16_t)ap[i];
#pragma unroll
            for (int i = 0; i < 8; ++i) hi[i] = (bf16_t)ap[8 + i];
            *(v8bf*)&Abuf[amsub * 512 + amrow * 16 + ajbase]        = lo;
            *(v8bf*)&Abuf[amsub * 512 + (amrow + 16) * 16 + ajbase] = hi;
        }
        // ---- stage B tile (32 x 128) from W[N,K] row-major, f32 -> bf16.
        //      Group-level uniform guards (N % 16 == 0 in all launches). ----
#pragma unroll
        for (int gsel = 0; gsel < 2; ++gsel) {
            const int g    = bg + gsel * 4;
            const int gnb  = n0 + g * 16;         // first column of this group
            v8bf blo, bhi;
            if (gnb + 16 <= N) {                  // fully in range (hot path)
#pragma unroll
                for (int j = 0; j < 8; ++j)
                    blo[j] = (bf16_t)W[(size_t)(gnb + (bk & 15)) * ldw + k0 + 8 * (bk >> 4) + j];
#pragma unroll
                for (int j = 0; j < 8; ++j)
                    bhi[j] = (bf16_t)W[(size_t)(gnb + (bk & 15)) * ldw + k0 + 16 + 8 * (bk >> 4) + j];
            } else {                              // fully out of range -> zeros
#pragma unroll
                for (int j = 0; j < 8; ++j) { blo[j] = (bf16_t)0.0f; bhi[j] = (bf16_t)0.0f; }
            }
            *(v8bf*)&Bbuf[g * 512 + bk * 16 + 0] = blo;
            *(v8bf*)&Bbuf[g * 512 + bk * 16 + 8] = bhi;
        }
        __syncthreads();

        if (k0 + 32 < K) {  // global_prefetch_b8 the next k-slabs
            __builtin_prefetch(A + (size_t)(m0 + am) * lda + (k0 + 32 + akg), 0, 1);
            __builtin_prefetch(W + (size_t)(n0 + bg * 16) * ldw + (k0 + 32 + bk), 0, 1);
        }

        v16bf afrag = *(const v16bf*)&Abuf[w * 512 + lane * 16];
#pragma unroll
        for (int g = 0; g < 8; ++g) {
            v16bf bfrag = *(const v16bf*)&Bbuf[g * 512 + lane * 16];
            // 8 args: (neg_a, A, neg_b, B, c_mod, C, reuse_a, reuse_b)
            acc[g] = __builtin_amdgcn_wmma_f32_16x16x32_bf16(
                false, afrag, false, bfrag, (short)0, acc[g], false, false);
        }
        __syncthreads();
    }

    // epilogue: C/D layout -> VGPR r: M = r + 8*(lane/16), N = lane%16
    const int half = lane >> 4;
    const int nlo  = lane & 15;
#pragma unroll
    for (int g = 0; g < 8; ++g) {
        const int gnb = n0 + g * 16;
        if (gnb < N) {                            // group uniform (N % 16 == 0)
            const int gn = gnb + nlo;
#pragma unroll
            for (int r = 0; r < 8; ++r) {
                int gm = m0 + w * 16 + r + 8 * half;
                C[(size_t)gm * ldc + gn] = acc[g][r];
            }
        }
    }
}

// =====================================================================
// Depthwise causal conv1d (4 taps) + SiLU:
//   u[b,l,d] = silu( sum_k conv_w[d,k] * xz_u[b, l-3+k, d] + conv_b[d] )
// xz is (NROW, 4096): u-half = cols [0,2048), z-half = cols [2048,4096)
// =====================================================================
__launch_bounds__(256)
__global__ void conv_silu_kernel(const float* __restrict__ xz,
                                 const float* __restrict__ conv_w,
                                 const float* __restrict__ conv_b,
                                 float* __restrict__ u)
{
    int tid = blockIdx.x * blockDim.x + threadIdx.x;
    if (tid >= NROW * D_INNER) return;
    int d    = tid & (D_INNER - 1);
    int row  = tid >> 11;            // b*SEQL + l
    int l    = row & (SEQL - 1);
    int base = row - l;              // b*SEQL
    float acc = conv_b[d];
#pragma unroll
    for (int k = 0; k < D_CONV; ++k) {
        int li = l - (D_CONV - 1) + k;
        if (li >= 0)
            acc += conv_w[d * D_CONV + k] *
                   xz[(size_t)(base + li) * (2 * D_INNER) + d];
    }
    u[(size_t)row * D_INNER + d] = acc * sigmoidf_(acc);
}

// =====================================================================
// Selective scan. One lane per (b, d, n): 2*2048*16 = 65536 lanes.
// Each 16-lane group shares a channel; y = sum_n state*C via 16-wide
// xor-butterfly (ds cross-lane). state lives in a register across L.
// Fuses: softplus(delta_raw + b_dt), deltaA = exp(delta*(-exp(A_log))),
// y += u*Dp, gate y *= z*sigmoid(z).
// =====================================================================
__launch_bounds__(256)
__global__ void scan_kernel(const float* __restrict__ xz,
                            const float* __restrict__ u,
                            const float* __restrict__ xdbl,   // (NROW, 96)
                            const float* __restrict__ draw,   // (NROW, D_INNER) pre-bias
                            const float* __restrict__ b_dt,
                            const float* __restrict__ A_log,
                            const float* __restrict__ Dp,
                            float* __restrict__ ybuf)
{
    int tid = blockIdx.x * blockDim.x + threadIdx.x;   // 65536 total
    int n   = tid & 15;
    int ch  = tid >> 4;                 // b*D_INNER + d
    int d   = ch & (D_INNER - 1);
    int b   = ch >> 11;

    const float Aval = -__expf(A_log[d * D_STATE + n]);
    const float bdt  = b_dt[d];
    const float Dpv  = Dp[d];
    float state = 0.0f;
    const int rowbase = b * SEQL;

    for (int l = 0; l < SEQL; ++l) {
        int row = rowbase + l;
        float dr    = draw[(size_t)row * D_INNER + d] + bdt;
        float delta = (dr > 20.0f) ? dr : log1pf(__expf(dr));
        float uv    = u[(size_t)row * D_INNER + d];
        float Bn    = xdbl[row * 96 + DT_RANK + n];
        float Cn    = xdbl[row * 96 + DT_RANK + D_STATE + n];

        state = __expf(delta * Aval) * state + delta * Bn * uv;
        float y = state * Cn;
        // 16-lane reduction (stays inside the channel's lane group)
        y += __shfl_xor(y, 8, 16);
        y += __shfl_xor(y, 4, 16);
        y += __shfl_xor(y, 2, 16);
        y += __shfl_xor(y, 1, 16);
        if (n == 0) {
            float zv = xz[(size_t)row * (2 * D_INNER) + D_INNER + d];
            float yy = y + uv * Dpv;
            yy *= zv * sigmoidf_(zv);          // y * silu(z)
            ybuf[(size_t)row * D_INNER + d] = yy;
        }
    }
}

// =====================================================================
// launch
// =====================================================================
extern "C" void kernel_launch(void* const* d_in, const int* in_sizes, int n_in,
                              void* d_out, int out_size, void* d_ws, size_t ws_size,
                              hipStream_t stream)
{
    const float* x      = (const float*)d_in[0];
    const float* W_in   = (const float*)d_in[1];
    const float* conv_w = (const float*)d_in[2];
    const float* conv_b = (const float*)d_in[3];
    const float* W_x    = (const float*)d_in[4];
    const float* W_dt   = (const float*)d_in[5];
    const float* b_dt   = (const float*)d_in[6];
    const float* A_log  = (const float*)d_in[7];
    const float* Dp     = (const float*)d_in[8];
    const float* W_out  = (const float*)d_in[9];
    float* out = (float*)d_out;

    // workspace layout (floats): ~84.7 MB total
    float* ws   = (float*)d_ws;
    float* xz   = ws;                                   // NROW * 4096
    float* u    = xz   + (size_t)NROW * 2 * D_INNER;    // NROW * 2048
    float* xdbl = u    + (size_t)NROW * D_INNER;        // NROW * 96
    float* draw = xdbl + (size_t)NROW * 96;             // NROW * 2048
    float* ybuf = draw + (size_t)NROW * D_INNER;        // NROW * 2048

    dim3 blk(128);

    // 1) xz = x @ W_in^T                (2048 x 4096, K=1024)
    gemm_wmma_bf16<<<dim3((2 * D_INNER) / 128, NROW / 64), blk, 0, stream>>>(
        x, W_in, xz, NROW, 2 * D_INNER, D_MODEL, D_MODEL, D_MODEL, 2 * D_INNER);

    // 2) depthwise causal conv + silu -> u
    conv_silu_kernel<<<(NROW * D_INNER) / 256, 256, 0, stream>>>(xz, conv_w, conv_b, u);

    // 3) x_dbl = u @ W_x^T              (2048 x 96, K=2048)
    gemm_wmma_bf16<<<dim3(1, NROW / 64), blk, 0, stream>>>(
        u, W_x, xdbl, NROW, DT_RANK + 2 * D_STATE, D_INNER, D_INNER, D_INNER, 96);

    // 4) draw = dt @ W_dt^T             (2048 x 2048, K=64); dt = x_dbl[:, :64]
    gemm_wmma_bf16<<<dim3(D_INNER / 128, NROW / 64), blk, 0, stream>>>(
        xdbl, W_dt, draw, NROW, D_INNER, DT_RANK, 96, DT_RANK, D_INNER);

    // 5) selective scan (+softplus, +u*Dp, +silu gate) -> ybuf
    scan_kernel<<<(BATCH * D_INNER * D_STATE) / 256, 256, 0, stream>>>(
        xz, u, xdbl, draw, b_dt, A_log, Dp, ybuf);

    // 6) out = ybuf @ W_out^T           (2048 x 1024, K=2048)
    gemm_wmma_bf16<<<dim3(D_MODEL / 128, NROW / 64), blk, 0, stream>>>(
        ybuf, W_out, out, NROW, D_MODEL, D_INNER, D_INNER, D_INNER, D_MODEL);
}
